// CrossAttentionDecoder_58815282151563
// MI455X (gfx1250) — matrix operand order repaired and checked by reading, not compiled, not run on hardware
//
#include <hip/hip_runtime.h>

// ---------------------------------------------------------------------------
// CrossAttentionDecoder on MI455X (gfx1250): ragged-segment flash attention
// with fused f16 K/V projection, all GEMMs on v_wmma_f32_16x16x32_f16,
// double-buffered async global->LDS K/V staging (ASYNCcnt).
// ---------------------------------------------------------------------------

typedef __attribute__((ext_vector_type(16))) _Float16 v16h;
typedef __attribute__((ext_vector_type(8)))  float    v8f;

#define NB     512
#define DIM    128
#define NLAT   16
#define NHEAD  4
#define HDIM   32
#define NLAYER 4
#define NDRUG  25600
#define NENZ   153600

__device__ __forceinline__ v8f wmma_f16(v16h a, v16h b, v8f c) {
  return __builtin_amdgcn_wmma_f32_16x16x32_f16(false, a, false, b, (short)0, c,
                                                false, false);
}

// A-fragment K index map for 16x32 f16 A: lanes 0-15 hold K={0..15},
// lanes 16-31 hold K={16..31} (contiguous, matching the B fragments).
__device__ __forceinline__ int a_k(int hi, int i) {
  return 16 * hi + i;
}

__device__ __forceinline__ v16h frag_a_f16(const _Float16* p, int ld) {
  int lane = threadIdx.x & 31;
  int m = lane & 15, hi = lane >> 4;
  const _Float16* row = p + (size_t)m * ld;
  v16h a;
#pragma unroll
  for (int i = 0; i < 16; ++i) a[i] = row[a_k(hi, i)];
  return a;
}

__device__ __forceinline__ v16h frag_a_f32(const float* p, int ld) {
  int lane = threadIdx.x & 31;
  int m = lane & 15, hi = lane >> 4;
  const float* row = p + (size_t)m * ld;
  v16h a;
#pragma unroll
  for (int i = 0; i < 16; ++i) a[i] = (_Float16)row[a_k(hi, i)];
  return a;
}

// B fragment for computing X @ W^T: logical B[k][n] = w[n][k], w row-major
// with leading dim ldk; caller passes w advanced to the (n0,k0) tile corner.
// Per lane: n = lane&15, 16 consecutive K -> 32B contiguous read.
__device__ __forceinline__ v16h frag_b_T(const _Float16* w, int ldk) {
  int lane = threadIdx.x & 31;
  int n = lane & 15, k0 = (lane >> 4) << 4;
  const _Float16* p = w + (size_t)n * ldk + k0;
  v16h b;
#pragma unroll
  for (int i = 0; i < 16; ++i) b[i] = p[i];
  return b;
}

// B fragment from row-major B[k][n] (leading dim ld), tile origin passed in.
__device__ __forceinline__ v16h frag_b(const _Float16* p, int ld) {
  int lane = threadIdx.x & 31;
  int n = lane & 15, k0 = (lane >> 4) << 4;
  v16h b;
#pragma unroll
  for (int i = 0; i < 16; ++i) b[i] = p[(size_t)(k0 + i) * ld + n];
  return b;
}

// ------------------------------- prep kernels -------------------------------

// seg[b] = lower_bound(batch, b); batch is sorted.
__global__ void seg_kernel(const int* __restrict__ batch, int nelem,
                           int* __restrict__ seg) {
  int b = blockIdx.x * blockDim.x + threadIdx.x;
  if (b > NB) return;
  int lo = 0, hi = nelem;
  while (lo < hi) {
    int mid = (lo + hi) >> 1;
    if (batch[mid] < b) lo = mid + 1; else hi = mid;
  }
  seg[b] = lo;
}

__global__ void cvt_kernel(const float* __restrict__ src,
                           _Float16* __restrict__ dst, long n) {
  long i = (long)blockIdx.x * blockDim.x + threadIdx.x;
  long stride = (long)gridDim.x * blockDim.x;
  for (; i < n; i += stride) dst[i] = (_Float16)src[i];
}

// Fold latent q-projection and MHA q-projection: Wc = Wqm @ Wq (f16),
// bc = Wqm @ bq + bqm (f32).
__global__ void foldq_kernel(const float* __restrict__ wq,
                             const float* __restrict__ bq,
                             const float* __restrict__ mw,
                             const float* __restrict__ mb,
                             _Float16* __restrict__ wc,
                             float* __restrict__ bc) {
  int idx = blockIdx.x * blockDim.x + threadIdx.x;
  if (idx >= NLAYER * DIM * DIM) return;
  int l = idx >> 14, j = (idx >> 7) & 127, i = idx & 127;
  const float* wqm = mw + (size_t)l * 3 * DIM * DIM;  // rows 0..127 = q proj
  const float* wql = wq + (size_t)l * DIM * DIM;
  float s = 0.f;
  for (int m = 0; m < DIM; ++m) s += wqm[j * DIM + m] * wql[m * DIM + i];
  wc[idx] = (_Float16)s;
  if (i == 0) {
    float t = mb[(size_t)l * 3 * DIM + j];
    for (int m = 0; m < DIM; ++m) t += wqm[j * DIM + m] * bq[l * DIM + m];
    bc[l * DIM + j] = t;
  }
}

__global__ void initlat_kernel(const float* __restrict__ latents,
                               float* __restrict__ lat) {
  int i = blockIdx.x * blockDim.x + threadIdx.x;
  if (i < NB * NLAT * DIM) lat[i] = latents[i & (NLAT * DIM - 1)];
}

// ------------------------------ per-layer: LN1+Q ----------------------------

__global__ __launch_bounds__(128)
void qproj_kernel(const float* __restrict__ lat,
                  const float* __restrict__ g, const float* __restrict__ bta,
                  const _Float16* __restrict__ wcd, const float* __restrict__ bcd,
                  const _Float16* __restrict__ wce, const float* __restrict__ bce,
                  _Float16* __restrict__ qhd, _Float16* __restrict__ qhe) {
  __shared__ _Float16 nl[NLAT * DIM];
  int b = blockIdx.x;
  const float* x = lat + (size_t)b * NLAT * DIM;
  if (threadIdx.x < NLAT) {
    int q = threadIdx.x;
    float s = 0.f;
    for (int k = 0; k < DIM; ++k) s += x[q * DIM + k];
    float mu = s * (1.f / DIM);
    float v = 0.f;
    for (int k = 0; k < DIM; ++k) { float d = x[q * DIM + k] - mu; v += d * d; }
    float inv = rsqrtf(v * (1.f / DIM) + 1e-5f);
    for (int k = 0; k < DIM; ++k)
      nl[q * DIM + k] = (_Float16)((x[q * DIM + k] - mu) * inv * g[k] + bta[k]);
  }
  __syncthreads();
  int wave = threadIdx.x >> 5, lane = threadIdx.x & 31;
  int n = lane & 15, hi = lane >> 4;
  // 16 N-tiles total (8 drug-q + 8 enzyme-q) over 4 waves
  for (int j = wave * 4; j < wave * 4 + 4; ++j) {
    int src = j >> 3, nt = j & 7;
    const _Float16* wc = src ? wce : wcd;
    const float* bc = src ? bce : bcd;
    _Float16* out = src ? qhe : qhd;
    v8f c = {};
#pragma unroll
    for (int ks = 0; ks < 4; ++ks)
      c = wmma_f16(frag_a_f16(nl + ks * 32, DIM),
                   frag_b_T(wc + (size_t)(nt * 16) * DIM + ks * 32, DIM), c);
    float bb = bc[nt * 16 + n];
#pragma unroll
    for (int r = 0; r < 8; ++r)
      out[(size_t)b * NLAT * DIM + (size_t)(r + 8 * hi) * DIM + nt * 16 + n] =
          (_Float16)(c[r] + bb);
  }
}

// -------------- per-layer: fused KV-projection + flash attention ------------
// One block per batch element, one wave per head. 32-key chunks, K/V staged by
// async global->LDS copies (ASYNCcnt) into a double buffer so the dominant
// memory stream overlaps the WMMA work:
//   kh/vh = chunk @ w^T (WMMA), S = q @ kh^T (WMMA), online softmax,
//   O += P @ vh (WMMA); epilogue: out-proj @ ow^T (WMMA).

__global__ __launch_bounds__(128)
void attn_kernel(const _Float16* __restrict__ kf, const _Float16* __restrict__ vf,
                 const int* __restrict__ seg, const _Float16* __restrict__ qh,
                 const _Float16* __restrict__ wk, const _Float16* __restrict__ wv,
                 const float* __restrict__ bk, const float* __restrict__ bv,
                 const _Float16* __restrict__ ow, const float* __restrict__ ob,
                 float* __restrict__ outb) {
  __shared__ __align__(16) _Float16 kbuf[2][32 * DIM];
  __shared__ __align__(16) _Float16 vbuf[2][32 * DIM];
  __shared__ __align__(16) _Float16 khs[NHEAD][32 * HDIM];
  __shared__ __align__(16) _Float16 vhs[NHEAD][32 * HDIM];
  __shared__ __align__(16) _Float16 ps[NHEAD][NLAT * 32];
  __shared__ __align__(16) _Float16 atile[NLAT * DIM];

  int b = blockIdx.x;
  int wave = threadIdx.x >> 5;
  int lane = threadIdx.x & 31;
  int n = lane & 15, hi = lane >> 4;
  int h = wave;

  int s0 = seg[b], s1 = seg[b + 1];
  int len = s1 - s0;
  int nch = (len + 31) >> 5;

  // Issue one 32x128 f16 K chunk + V chunk into buffer (c&1) as 8 async
  // global->LDS b128 copies per thread-slot sweep. Rows past the segment end
  // load a clamped (valid) address so every wave's ASYNCcnt advances
  // uniformly; the tail is zeroed after the wait.
  auto issue = [&](int c) {
    int bsel = c & 1;
#pragma unroll
    for (int s = 0; s < 4; ++s) {
      int t = threadIdx.x + s * 128;
      int r = t >> 4, off = (t & 15) << 3;          // 8 halves = 16 bytes
      int rr = c * 32 + r; if (rr > len - 1) rr = len - 1;
      size_t gidx = (size_t)(s0 + rr) * DIM + off;
      unsigned lk = (unsigned)(uintptr_t)(&kbuf[bsel][r * DIM + off]);
      unsigned lv = (unsigned)(uintptr_t)(&vbuf[bsel][r * DIM + off]);
      asm volatile("global_load_async_to_lds_b128 %0, %1, off"
                   :: "v"(lk), "v"(kf + gidx) : "memory");
      asm volatile("global_load_async_to_lds_b128 %0, %1, off"
                   :: "v"(lv), "v"(vf + gidx) : "memory");
    }
  };

  // Per-head query fragment, pre-scaled by 1/sqrt(HDIM).
  v16h qfrag;
  {
    const _Float16* qp = qh + (size_t)b * NLAT * DIM + h * HDIM;
    int m = lane & 15;
#pragma unroll
    for (int i = 0; i < 16; ++i)
      qfrag[i] = (_Float16)((float)qp[m * DIM + a_k(hi, i)] * 0.1767766952966369f);
  }

  float bk0 = bk[h * HDIM + n],      bk1 = bk[h * HDIM + 16 + n];
  float bv0 = bv[h * HDIM + n],      bv1 = bv[h * HDIM + 16 + n];

  float rowm[8], rowl[8];
  v8f o0 = {}, o1 = {};
#pragma unroll
  for (int r = 0; r < 8; ++r) { rowm[r] = -1e30f; rowl[r] = 0.f; }

  if (len > 0) issue(0);

  for (int c = 0; c < nch; ++c) {
    int bsel = c & 1;
    int base = c * 32;
    int rem = len - base;
    if (c + 1 < nch) {
      issue(c + 1);                                   // prefetch into other buf
      asm volatile("s_wait_asynccnt 8" ::: "memory"); // chunk c complete
    } else {
      asm volatile("s_wait_asynccnt 0" ::: "memory");
    }
    if (rem < 32) {  // zero tail rows (keeps NaN-free data out of P@V WMMA)
      for (int t = threadIdx.x; t < 512; t += 128) {
        int r = t >> 4, off = (t & 15) << 3;
        if (r >= rem) {
          uint4 z = {0, 0, 0, 0};
          *(uint4*)(&kbuf[bsel][r * DIM + off]) = z;
          *(uint4*)(&vbuf[bsel][r * DIM + off]) = z;
        }
      }
    }
    __syncthreads();

    // Project this head's K/V slice: (32 x 128) @ w[h*32 .. h*32+32)^T + bias.
#pragma unroll
    for (int mt = 0; mt < 2; ++mt) {
      v8f ck0 = {}, ck1 = {}, cv0 = {}, cv1 = {};
#pragma unroll
      for (int ks = 0; ks < 4; ++ks) {
        v16h ak = frag_a_f16(&kbuf[bsel][mt * 16 * DIM + ks * 32], DIM);
        ck0 = wmma_f16(ak, frag_b_T(wk + (size_t)(h * HDIM) * DIM + ks * 32, DIM), ck0);
        ck1 = wmma_f16(ak, frag_b_T(wk + (size_t)(h * HDIM + 16) * DIM + ks * 32, DIM), ck1);
        v16h av = frag_a_f16(&vbuf[bsel][mt * 16 * DIM + ks * 32], DIM);
        cv0 = wmma_f16(av, frag_b_T(wv + (size_t)(h * HDIM) * DIM + ks * 32, DIM), cv0);
        cv1 = wmma_f16(av, frag_b_T(wv + (size_t)(h * HDIM + 16) * DIM + ks * 32, DIM), cv1);
      }
#pragma unroll
      for (int r = 0; r < 8; ++r) {
        int key = mt * 16 + r + 8 * hi;
        khs[h][key * HDIM + n]      = (_Float16)(ck0[r] + bk0);
        khs[h][key * HDIM + 16 + n] = (_Float16)(ck1[r] + bk1);
        vhs[h][key * HDIM + n]      = (_Float16)(cv0[r] + bv0);
        vhs[h][key * HDIM + 16 + n] = (_Float16)(cv1[r] + bv1);
      }
    }
    asm volatile("s_wait_dscnt 0" ::: "memory");  // same-wave LDS RAW ordering

    // Scores: S = q (16x32) @ kh^T -> two 16x16 f32 tiles.
    v8f S0, S1;
    {
      v8f z = {};
      S0 = wmma_f16(qfrag, frag_b_T(khs[h], HDIM), z);
      S1 = wmma_f16(qfrag, frag_b_T(khs[h] + 16 * HDIM, HDIM), z);
    }
    bool bad0 = (base + n) >= len;
    bool bad1 = (base + 16 + n) >= len;

    // Online softmax. Row m = r + 8*hi lives on 16 lanes of a half-wave;
    // shfl_xor over masks {1,2,4,8} reduces within that half.
#pragma unroll
    for (int r = 0; r < 8; ++r) {
      float x0 = bad0 ? -1e9f : S0[r];
      float x1 = bad1 ? -1e9f : S1[r];
      float t = fmaxf(x0, x1);
      t = fmaxf(t, __shfl_xor(t, 1));
      t = fmaxf(t, __shfl_xor(t, 2));
      t = fmaxf(t, __shfl_xor(t, 4));
      t = fmaxf(t, __shfl_xor(t, 8));
      float mnew = fmaxf(rowm[r], t);
      float alpha = __expf(rowm[r] - mnew);
      float p0 = __expf(x0 - mnew);
      float p1 = __expf(x1 - mnew);
      float ls = p0 + p1;
      ls += __shfl_xor(ls, 1);
      ls += __shfl_xor(ls, 2);
      ls += __shfl_xor(ls, 4);
      ls += __shfl_xor(ls, 8);
      rowm[r] = mnew;
      rowl[r] = rowl[r] * alpha + ls;
      o0[r] *= alpha;
      o1[r] *= alpha;
      int q = r + 8 * hi;
      ps[h][q * 32 + n]      = (_Float16)p0;
      ps[h][q * 32 + 16 + n] = (_Float16)p1;
    }
    asm volatile("s_wait_dscnt 0" ::: "memory");

    // O += P (16x32) @ vh (32 x 32)
    v16h pf = frag_a_f16(ps[h], 32);
    o0 = wmma_f16(pf, frag_b(vhs[h], HDIM), o0);
    o1 = wmma_f16(pf, frag_b(vhs[h] + 16, HDIM), o1);

    __syncthreads();  // all waves done with buf[c&1] before it is re-issued
  }

  // Normalize, gather heads into one 16x128 f16 tile.
#pragma unroll
  for (int r = 0; r < 8; ++r) {
    int q = r + 8 * hi;
    float inv = rowl[r] > 0.f ? 1.f / rowl[r] : 0.f;
    atile[q * DIM + h * HDIM + n]      = (_Float16)(o0[r] * inv);
    atile[q * DIM + h * HDIM + 16 + n] = (_Float16)(o1[r] * inv);
  }
  __syncthreads();

  // Output projection: (16x128) @ ow^T + ob -> f32 contribution buffer.
  for (int j = 0; j < 2; ++j) {
    int nt = wave * 2 + j;
    v8f c = {};
#pragma unroll
    for (int ks = 0; ks < 4; ++ks)
      c = wmma_f16(frag_a_f16(atile + ks * 32, DIM),
                   frag_b_T(ow + (size_t)(nt * 16) * DIM + ks * 32, DIM), c);
    float bb = ob[nt * 16 + n];
#pragma unroll
    for (int r = 0; r < 8; ++r) {
      int q = r + 8 * hi;
      outb[(size_t)b * NLAT * DIM + (size_t)q * DIM + nt * 16 + n] = c[r] + bb;
    }
  }
}

// -------------------- per-layer: residual + LN2 + FFN -----------------------

__global__ __launch_bounds__(128)
void ffn_kernel(float* __restrict__ lat,
                const float* __restrict__ ad, const float* __restrict__ ae,
                const float* __restrict__ g, const float* __restrict__ bta,
                const _Float16* __restrict__ w1, const float* __restrict__ b1,
                const _Float16* __restrict__ w2, const float* __restrict__ b2) {
  __shared__ float xm[NLAT * DIM];
  __shared__ _Float16 nl[NLAT * DIM];
  __shared__ _Float16 h1[NLAT * 2 * DIM];
  int b = blockIdx.x;
  size_t off = (size_t)b * NLAT * DIM;
  for (int t = threadIdx.x; t < NLAT * DIM; t += 128)
    xm[t] = lat[off + t] + ad[off + t] + ae[off + t];
  __syncthreads();
  if (threadIdx.x < NLAT) {
    int q = threadIdx.x;
    float s = 0.f;
    for (int k = 0; k < DIM; ++k) s += xm[q * DIM + k];
    float mu = s * (1.f / DIM);
    float v = 0.f;
    for (int k = 0; k < DIM; ++k) { float d = xm[q * DIM + k] - mu; v += d * d; }
    float inv = rsqrtf(v * (1.f / DIM) + 1e-5f);
    for (int k = 0; k < DIM; ++k)
      nl[q * DIM + k] = (_Float16)((xm[q * DIM + k] - mu) * inv * g[k] + bta[k]);
  }
  __syncthreads();
  int wave = threadIdx.x >> 5, lane = threadIdx.x & 31;
  int n = lane & 15, hi = lane >> 4;
  // GEMM1: 16 x 256 = 16 N-tiles over 4 waves, K = 128
  for (int j = 0; j < 4; ++j) {
    int nt = wave * 4 + j;
    v8f c = {};
#pragma unroll
    for (int ks = 0; ks < 4; ++ks)
      c = wmma_f16(frag_a_f16(nl + ks * 32, DIM),
                   frag_b_T(w1 + (size_t)(nt * 16) * DIM + ks * 32, DIM), c);
    float bb = b1[nt * 16 + n];
#pragma unroll
    for (int r = 0; r < 8; ++r)
      h1[(r + 8 * hi) * (2 * DIM) + nt * 16 + n] = (_Float16)fmaxf(c[r] + bb, 0.f);
  }
  __syncthreads();
  // GEMM2: 16 x 128, K = 256; add residual and store lat.
  for (int j = 0; j < 2; ++j) {
    int nt = wave * 2 + j;
    v8f c = {};
#pragma unroll
    for (int ks = 0; ks < 8; ++ks)
      c = wmma_f16(frag_a_f16(h1 + ks * 32, 2 * DIM),
                   frag_b_T(w2 + (size_t)(nt * 16) * (2 * DIM) + ks * 32, 2 * DIM), c);
    float bb = b2[nt * 16 + n];
#pragma unroll
    for (int r = 0; r < 8; ++r) {
      int q = r + 8 * hi;
      lat[off + (size_t)q * DIM + nt * 16 + n] = xm[q * DIM + nt * 16 + n] + c[r] + bb;
    }
  }
}

// ------------------------------- head kernel --------------------------------

__global__ __launch_bounds__(128)
void head_kernel(const float* __restrict__ lat,
                 const _Float16* __restrict__ hw1, const float* __restrict__ hb1,
                 const float* __restrict__ hw2, const float* __restrict__ hb2,
                 float* __restrict__ out) {
  __shared__ float hrow[NLAT * DIM];
  int b0 = blockIdx.x * 16;
  int wave = threadIdx.x >> 5, lane = threadIdx.x & 31;
  int n = lane & 15, hi = lane >> 4;
  const int KK = NLAT * DIM;  // 2048
  for (int j = 0; j < 2; ++j) {
    int nt = wave * 2 + j;
    v8f c = {};
    for (int ks = 0; ks < KK / 32; ++ks)
      c = wmma_f16(frag_a_f32(lat + (size_t)b0 * KK + ks * 32, KK),
                   frag_b_T(hw1 + (size_t)(nt * 16) * KK + ks * 32, KK), c);
    float bb = hb1[nt * 16 + n];
#pragma unroll
    for (int r = 0; r < 8; ++r)
      hrow[(r + 8 * hi) * DIM + nt * 16 + n] = fmaxf(c[r] + bb, 0.f);
  }
  __syncthreads();
  if (threadIdx.x < 16) {
    int m = threadIdx.x;
    float s = hb2[0];
    for (int k = 0; k < DIM; ++k) s += hrow[m * DIM + k] * hw2[k];
    out[b0 + m] = (s > 20.f) ? s : log1pf(__expf(s));
  }
}

// ---------------------------------------------------------------------------

extern "C" void kernel_launch(void* const* d_in, const int* in_sizes, int n_in,
                              void* d_out, int out_size, void* d_ws, size_t ws_size,
                              hipStream_t stream) {
  (void)in_sizes; (void)n_in; (void)out_size; (void)ws_size;
  const float* drug_k   = (const float*)d_in[0];
  const float* drug_v   = (const float*)d_in[1];
  const float* enz_k    = (const float*)d_in[2];
  const float* enz_v    = (const float*)d_in[3];
  const int*   drug_b   = (const int*)d_in[4];
  const int*   enz_b    = (const int*)d_in[5];
  const float* latents  = (const float*)d_in[6];
  const float* wq_d     = (const float*)d_in[7];
  const float* bq_d     = (const float*)d_in[8];
  const float* wq_e     = (const float*)d_in[9];
  const float* bq_e     = (const float*)d_in[10];
  const float* mha_d_w  = (const float*)d_in[11];
  const float* mha_d_b  = (const float*)d_in[12];
  const float* mha_d_ow = (const float*)d_in[13];
  const float* mha_d_ob = (const float*)d_in[14];
  const float* mha_e_w  = (const float*)d_in[15];
  const float* mha_e_b  = (const float*)d_in[16];
  const float* mha_e_ow = (const float*)d_in[17];
  const float* mha_e_ob = (const float*)d_in[18];
  const float* ln1_g    = (const float*)d_in[19];
  const float* ln1_b    = (const float*)d_in[20];
  const float* ln2_g    = (const float*)d_in[21];
  const float* ln2_b    = (const float*)d_in[22];
  const float* ffn_w1   = (const float*)d_in[23];
  const float* ffn_b1   = (const float*)d_in[24];
  const float* ffn_w2   = (const float*)d_in[25];
  const float* ffn_b2   = (const float*)d_in[26];
  const float* head_w1  = (const float*)d_in[27];
  const float* head_b1  = (const float*)d_in[28];
  const float* head_w2  = (const float*)d_in[29];
  const float* head_b2  = (const float*)d_in[30];

  char* base = (char*)d_ws;
  size_t off = 0;
  auto carve = [&](size_t bytes) -> void* {
    void* p = base + off;
    off = (off + bytes + 255) & ~(size_t)255;
    return p;
  };
  float*    lat  = (float*)carve(sizeof(float) * NB * NLAT * DIM);
  float*    ad   = (float*)carve(sizeof(float) * NB * NLAT * DIM);
  float*    ae   = (float*)carve(sizeof(float) * NB * NLAT * DIM);
  _Float16* qhd  = (_Float16*)carve(2L * NB * NLAT * DIM);
  _Float16* qhe  = (_Float16*)carve(2L * NB * NLAT * DIM);
  int*      segd = (int*)carve(4 * (NB + 1));
  int*      sege = (int*)carve(4 * (NB + 1));
  _Float16* wcd  = (_Float16*)carve(2L * NLAYER * DIM * DIM);
  _Float16* wce  = (_Float16*)carve(2L * NLAYER * DIM * DIM);
  float*    bcd  = (float*)carve(4L * NLAYER * DIM);
  float*    bce  = (float*)carve(4L * NLAYER * DIM);
  _Float16* wkd  = (_Float16*)carve(2L * NLAYER * DIM * DIM);
  _Float16* wvd  = (_Float16*)carve(2L * NLAYER * DIM * DIM);
  _Float16* wke  = (_Float16*)carve(2L * NLAYER * DIM * DIM);
  _Float16* wve  = (_Float16*)carve(2L * NLAYER * DIM * DIM);
  _Float16* owd  = (_Float16*)carve(2L * NLAYER * DIM * DIM);
  _Float16* owe  = (_Float16*)carve(2L * NLAYER * DIM * DIM);
  _Float16* w1h  = (_Float16*)carve(2L * NLAYER * 2 * DIM * DIM);
  _Float16* w2h  = (_Float16*)carve(2L * NLAYER * 2 * DIM * DIM);
  _Float16* hw1h = (_Float16*)carve(2L * DIM * NLAT * DIM);
  _Float16* kd16 = (_Float16*)carve(2L * NDRUG * DIM);
  _Float16* vd16 = (_Float16*)carve(2L * NDRUG * DIM);
  _Float16* ke16 = (_Float16*)carve(2L * NENZ * DIM);
  _Float16* ve16 = (_Float16*)carve(2L * NENZ * DIM);

  // -------- one-time prep --------
  seg_kernel<<<3, 256, 0, stream>>>(drug_b, NDRUG, segd);
  seg_kernel<<<3, 256, 0, stream>>>(enz_b, NENZ, sege);

  for (int l = 0; l < NLAYER; ++l) {
    size_t wb = (size_t)l * 3 * DIM * DIM, wo = (size_t)l * DIM * DIM;
    cvt_kernel<<<64, 256, 0, stream>>>(mha_d_w + wb + DIM * DIM,     wkd + wo, DIM * DIM);
    cvt_kernel<<<64, 256, 0, stream>>>(mha_d_w + wb + 2 * DIM * DIM, wvd + wo, DIM * DIM);
    cvt_kernel<<<64, 256, 0, stream>>>(mha_e_w + wb + DIM * DIM,     wke + wo, DIM * DIM);
    cvt_kernel<<<64, 256, 0, stream>>>(mha_e_w + wb + 2 * DIM * DIM, wve + wo, DIM * DIM);
  }
  cvt_kernel<<<256, 256, 0, stream>>>(mha_d_ow, owd, (long)NLAYER * DIM * DIM);
  cvt_kernel<<<256, 256, 0, stream>>>(mha_e_ow, owe, (long)NLAYER * DIM * DIM);
  cvt_kernel<<<256, 256, 0, stream>>>(ffn_w1, w1h, (long)NLAYER * 2 * DIM * DIM);
  cvt_kernel<<<256, 256, 0, stream>>>(ffn_w2, w2h, (long)NLAYER * 2 * DIM * DIM);
  cvt_kernel<<<256, 256, 0, stream>>>(head_w1, hw1h, (long)DIM * NLAT * DIM);
  cvt_kernel<<<2048, 256, 0, stream>>>(drug_k, kd16, (long)NDRUG * DIM);
  cvt_kernel<<<2048, 256, 0, stream>>>(drug_v, vd16, (long)NDRUG * DIM);
  cvt_kernel<<<4096, 256, 0, stream>>>(enz_k, ke16, (long)NENZ * DIM);
  cvt_kernel<<<4096, 256, 0, stream>>>(enz_v, ve16, (long)NENZ * DIM);

  foldq_kernel<<<NLAYER * DIM * DIM / 256, 256, 0, stream>>>(wq_d, bq_d, mha_d_w,
                                                             mha_d_b, wcd, bcd);
  foldq_kernel<<<NLAYER * DIM * DIM / 256, 256, 0, stream>>>(wq_e, bq_e, mha_e_w,
                                                             mha_e_b, wce, bce);
  initlat_kernel<<<NB * NLAT * DIM / 256, 256, 0, stream>>>(latents, lat);

  // -------- layers --------
  for (int l = 0; l < NLAYER; ++l) {
    size_t wo = (size_t)l * DIM * DIM;
    qproj_kernel<<<NB, 128, 0, stream>>>(lat, ln1_g + l * DIM, ln1_b + l * DIM,
                                         wcd + wo, bcd + l * DIM,
                                         wce + wo, bce + l * DIM, qhd, qhe);
    attn_kernel<<<NB, 128, 0, stream>>>(kd16, vd16, segd, qhd, wkd + wo, wvd + wo,
                                        mha_d_b + (size_t)l * 3 * DIM + DIM,
                                        mha_d_b + (size_t)l * 3 * DIM + 2 * DIM,
                                        owd + wo, mha_d_ob + l * DIM, ad);
    attn_kernel<<<NB, 128, 0, stream>>>(ke16, ve16, sege, qhe, wke + wo, wve + wo,
                                        mha_e_b + (size_t)l * 3 * DIM + DIM,
                                        mha_e_b + (size_t)l * 3 * DIM + 2 * DIM,
                                        owe + wo, mha_e_ob + l * DIM, ae);
    ffn_kernel<<<NB, 128, 0, stream>>>(lat, ad, ae, ln2_g + l * DIM, ln2_b + l * DIM,
                                       w1h + (size_t)l * 2 * DIM * DIM,
                                       ffn_b1 + l * 2 * DIM,
                                       w2h + (size_t)l * 2 * DIM * DIM,
                                       ffn_b2 + l * DIM);
  }
  head_kernel<<<NB / 16, 128, 0, stream>>>(lat, hw1h, head_b1, head_w2, head_b2,
                                           (float*)d_out);
}
